// Reformer_76416058131365
// MI455X (gfx1250) — matrix operand order repaired and checked by reading, not compiled, not run on hardware
//
#include <hip/hip_runtime.h>
#include <hip/hip_bf16.h>
#include <math.h>

// ---------------- problem dims ----------------
#define Bsz     2
#define Tsz     4096
#define Esz     512
#define Hn      8
#define DHn     64          // E / H
#define DEPTHn  4
#define NHASHn  4
#define NBn     64
#define BUCKETn 64
#define FFn     2048
#define NCLASSn 10
#define BHn     (Bsz*Hn)        // 16
#define NRn     (NHASHn*Tsz)    // 16384 sorted rows per b
#define Cn      (NHASHn*NBn)    // 256 chunks per b

typedef __attribute__((ext_vector_type(16))) _Float16 v16h;
typedef __attribute__((ext_vector_type(8)))  float    v8f;

// LDS byte address of a __shared__ pointer (generic LDS aperture addr[31:0] is
// the wave-relative LDS offset per ISA 10.2).
__device__ __forceinline__ unsigned lds_addr_of(const void* p) {
    return (unsigned)(uintptr_t)p;
}
// Async copy 16B global -> LDS, tracked by ASYNCcnt (CDNA5).
__device__ __forceinline__ void async_copy_b128(unsigned lds, const void* gptr) {
    asm volatile("global_load_async_to_lds_b128 %0, %1, off"
                 :: "v"(lds), "v"((unsigned long long)gptr) : "memory");
}
__device__ __forceinline__ void wait_async0() {
    asm volatile("s_wait_asynccnt 0" ::: "memory");
}

// ---------------- WMMA fragment loaders (contiguous K order, wave32) -------
// A 16x32 f16: lanes 0-15 -> M=lane, K = 0..15; lanes 16-31 -> M=lane-16,
// K = 16..31.  base points at the 32-wide K slice; ld = row stride.
__device__ __forceinline__ v16h fragA(const _Float16* base, int ld, int row, int lane) {
    const int kb = (lane & 16) ? 16 : 0;
    const _Float16* p = base + row * ld + kb;
    v16h f;
#pragma unroll
    for (int e = 0; e < 16; ++e) f[e] = p[e];
    return f;
}
// B 32x16 f16 stored K-transposed in LDS: baseT[n*ld + k].  lanes 0-15 -> N=lane,
// K=0..15; lanes 16-31 -> N=lane-16, K=16..31 (contiguous run per lane).
__device__ __forceinline__ v16h fragB(const _Float16* baseT, int ld, int ncol, int lane) {
    const int n  = ncol + (lane & 15);
    const int kb = (lane & 16) ? 16 : 0;
    const _Float16* p = baseT + n * ld + kb;
    v16h f;
#pragma unroll
    for (int e = 0; e < 16; ++e) f[e] = p[e];
    return f;
}

// ---------------- embedding ----------------
__global__ void __launch_bounds__(256)
embed_kernel(const int* __restrict__ xi, const float* __restrict__ tok,
             const float* __restrict__ pos, float* __restrict__ x1, float* __restrict__ x2) {
    const size_t rt = blockIdx.x;            // over B*T
    const int t = (int)(rt % Tsz);
    const int v = xi[rt];
    for (int e = threadIdx.x; e < Esz; e += 256) {
        float val = tok[(size_t)v * Esz + e] + pos[(size_t)t * Esz + e];
        x1[rt * Esz + e] = val;
        x2[rt * Esz + e] = val;
    }
}

// ---------------- layernorm (f32 in, f16 out) ----------------
__global__ void __launch_bounds__(256)
layernorm_to_f16(const float* __restrict__ X, const float* __restrict__ g,
                 const float* __restrict__ be, _Float16* __restrict__ Y) {
    __shared__ float red[256];
    const size_t row = blockIdx.x;
    const int tid = threadIdx.x;
    float a0 = X[row * Esz + tid];
    float a1 = X[row * Esz + 256 + tid];
    red[tid] = a0 + a1;
    __syncthreads();
    for (int s = 128; s > 0; s >>= 1) { if (tid < s) red[tid] += red[tid + s]; __syncthreads(); }
    float mean = red[0] * (1.f / Esz);
    __syncthreads();
    red[tid] = (a0 - mean) * (a0 - mean) + (a1 - mean) * (a1 - mean);
    __syncthreads();
    for (int s = 128; s > 0; s >>= 1) { if (tid < s) red[tid] += red[tid + s]; __syncthreads(); }
    float inv = rsqrtf(red[0] * (1.f / Esz) + 1e-5f);
    Y[row * Esz + tid]       = (_Float16)((a0 - mean) * inv * g[tid] + be[tid]);
    Y[row * Esz + 256 + tid] = (_Float16)((a1 - mean) * inv * g[256 + tid] + be[256 + tid]);
}

// ---------------- weight convert + transpose: f32 (KxN) -> f16 (NxK) -------
__global__ void cvt_transpose_f16(const float* __restrict__ in, _Float16* __restrict__ out,
                                  int K, int N) {
    int i = blockIdx.x * 256 + threadIdx.x;     // over K*N, coalesced reads
    if (i >= K * N) return;
    int k = i / N, n = i - k * N;
    out[(size_t)n * K + k] = (_Float16)in[i];
}

// ---------------- tiled WMMA GEMM: C(MxN) = A(MxK) * B(KxN), B given as B^T --
// Both A (64x32) and B^T (128x32) tiles staged via async-LDS loads, double-
// buffered.  Inner loop: ds_load_b128 fragments -> 8 WMMA / wave / K-step.
// epilogue: (+bias[N]) -> (gelu) -> (+resid) -> write f32 and/or f16
__global__ void __launch_bounds__(128)
gemm_f16_wmma(const _Float16* __restrict__ A, const _Float16* __restrict__ BT,
              float* __restrict__ Cf, _Float16* __restrict__ Ch,
              const float* __restrict__ bias, const float* __restrict__ resid,
              int M, int N, int K, int fuseGelu) {
    __shared__ _Float16 As[2][64 * 32];
    __shared__ _Float16 Bs[2][128 * 32];   // Bs[n*32 + k], n relative to col0
    const int row0 = blockIdx.y * 64;
    const int col0 = blockIdx.x * 128;
    const int tid = threadIdx.x;
    const int wave = tid >> 5, lane = tid & 31;

    // A tile 64x32: 256 x 16B chunks; chunk c -> LDS byte 16*c, row c>>2, col (c&3)*8
    auto stageA = [&](int kk, int buf) {
        const unsigned base = lds_addr_of(&As[buf][0]);
#pragma unroll
        for (int c = tid; c < 256; c += 128) {
            int r = c >> 2, cc = (c & 3) << 3;
            async_copy_b128(base + (unsigned)c * 16u,
                            A + (size_t)(row0 + r) * K + kk + cc);
        }
    };
    // B^T tile 128x32: 512 x 16B chunks, same scheme (contiguous along K)
    auto stageB = [&](int kk, int buf) {
        const unsigned base = lds_addr_of(&Bs[buf][0]);
#pragma unroll
        for (int c = tid; c < 512; c += 128) {
            int r = c >> 2, cc = (c & 3) << 3;
            async_copy_b128(base + (unsigned)c * 16u,
                            BT + (size_t)(col0 + r) * K + kk + cc);
        }
    };

    const v8f vz = {0, 0, 0, 0, 0, 0, 0, 0};
    v8f acc[8];
#pragma unroll
    for (int j = 0; j < 8; ++j) acc[j] = vz;

    stageA(0, 0);
    stageB(0, 0);
    wait_async0();
    __syncthreads();

    int buf = 0;
    for (int kk = 0; kk < K; kk += 32) {
        const int nxt = buf ^ 1;
        if (kk + 32 < K) {          // prefetch next K slice into alternate buffer
            stageA(kk + 32, nxt);
            stageB(kk + 32, nxt);
        }
        v16h a = fragA(As[buf], 32, 16 * wave + (lane & 15), lane);
#pragma unroll
        for (int j = 0; j < 8; ++j) {
            v16h bf = fragB(Bs[buf], 32, 16 * j, lane);
            acc[j] = __builtin_amdgcn_wmma_f32_16x16x32_f16(false, a, false, bf,
                                                            (short)0, acc[j], false, false);
        }
        wait_async0();
        __syncthreads();
        buf = nxt;
    }

    const int nl = lane & 15;
    const int mhi = (lane & 16) ? 8 : 0;
#pragma unroll
    for (int j = 0; j < 8; ++j) {
        int col = col0 + 16 * j + nl;
        float bv = bias ? bias[col] : 0.f;
#pragma unroll
        for (int r = 0; r < 8; ++r) {
            int row = row0 + 16 * wave + mhi + r;
            float v = acc[j][r] + bv;
            if (fuseGelu) v = 0.5f * v * (1.f + erff(v * 0.70710678118f));
            if (resid) v += resid[(size_t)row * N + col];
            if (Cf) Cf[(size_t)row * N + col] = v;
            if (Ch) Ch[(size_t)row * N + col] = (_Float16)v;
        }
    }
}

// ---------------- LSH hashing: keys = bucket*T + t ----------------
__global__ void __launch_bounds__(256)
lsh_hash(const float* __restrict__ qk, const float* __restrict__ rot, int* __restrict__ keys) {
    int gid = blockIdx.x * 256 + threadIdx.x;
    if (gid >= BHn * NHASHn * Tsz) return;
    int b = gid / (NHASHn * Tsz);
    int rem = gid - b * (NHASHn * Tsz);
    int nh = rem / Tsz;
    int t = rem - nh * Tsz;
    int bb = b / Hn, hh = b - bb * Hn;
    const float* q = qk + ((size_t)bb * Tsz + t) * Esz + hh * DHn;
    float r[NBn / 2];
    for (int i = 0; i < NBn / 2; ++i) {
        float s = 0.f;
        for (int f = 0; f < DHn; ++f)
            s += q[f] * rot[((size_t)f * NHASHn + nh) * (NBn / 2) + i];
        r[i] = s;
    }
    float best = -3.4e38f; int bi = 0;
    for (int i = 0; i < NBn; ++i) {              // argmax over [r, -r], first-wins
        float v = (i < NBn / 2) ? r[i] : -r[i - NBn / 2];
        if (v > best) { best = v; bi = i; }
    }
    int bucket = bi + nh * NBn;
    keys[(size_t)b * NRn + rem] = bucket * Tsz + t;
}

// ---------------- bitonic sort of 16384 (key, idx) per b in LDS ----------------
__global__ void __launch_bounds__(1024)
bitonic_sort(const int* __restrict__ keys, int* __restrict__ sticker, int* __restrict__ undo) {
    extern __shared__ int sm[];
    int* kS = sm;
    int* vS = sm + NRn;
    const int b = blockIdx.x;
    const int tid = threadIdx.x;
    for (int i = tid; i < NRn; i += 1024) { kS[i] = keys[(size_t)b * NRn + i]; vS[i] = i; }
    __syncthreads();
    for (int ksz = 2; ksz <= NRn; ksz <<= 1) {
        for (int j = ksz >> 1; j > 0; j >>= 1) {
            for (int p = 0; p < NRn / 1024; ++p) {
                int i = tid + (p << 10);
                int ixj = i ^ j;
                if (ixj > i) {
                    bool up = ((i & ksz) == 0);
                    int ki = kS[i], kj = kS[ixj];
                    if ((ki > kj) == up) {
                        kS[i] = kj; kS[ixj] = ki;
                        int tv = vS[i]; vS[i] = vS[ixj]; vS[ixj] = tv;
                    }
                }
            }
            __syncthreads();
        }
    }
    for (int i = tid; i < NRn; i += 1024) {
        int s = vS[i];
        sticker[(size_t)b * NRn + i] = s;
        undo[(size_t)b * NRn + s] = i;
    }
}

// ---------------- gather sorted qk / v into f16 ----------------
__global__ void __launch_bounds__(256)
gather_sorted(const int* __restrict__ sticker, const float* __restrict__ qk,
              const float* __restrict__ v, _Float16* __restrict__ sqk,
              _Float16* __restrict__ sv) {
    size_t rid = (size_t)blockIdx.x * 4 + (threadIdx.x >> 6);   // b*NRn + pos
    int e = threadIdx.x & 63;
    int b = (int)(rid / NRn);
    int t = sticker[rid] % Tsz;
    int bb = b / Hn, hh = b - bb * Hn;
    size_t src = ((size_t)bb * Tsz + t) * Esz + (size_t)hh * DHn + e;
    sqk[rid * 64 + e] = (_Float16)qk[src];
    sv[rid * 64 + e]  = (_Float16)v[src];
}

// ---------------- chunked LSH attention (WMMA) ----------------
__global__ void __launch_bounds__(128)
lsh_attn_chunk(const _Float16* __restrict__ sqk, const _Float16* __restrict__ sv,
               const int* __restrict__ sticker, float* __restrict__ so,
               float* __restrict__ slse) {
    extern __shared__ char smem[];
    _Float16* qS = (_Float16*)smem;                         // 64x64
    _Float16* kS = (_Float16*)(smem + 8192);                // 128x64; reused as probs 64x128
    _Float16* vT = (_Float16*)(smem + 8192 + 16384);        // vT[dh][j] 64x128
    float*    dS = (float*)(smem + 8192 + 16384 + 16384);   // 64x128
    int* qtS = (int*)(smem + 8192 + 16384 + 16384 + 32768); // 64
    int* ktS = qtS + 64;                                    // 128
    const int c = blockIdx.x;
    const int b = blockIdx.y;
    const int cp = (c + Cn - 1) % Cn;
    const int tid = threadIdx.x;
    const int wave = tid >> 5, lane = tid & 31;
    const size_t baseQ = (size_t)b * NRn + (size_t)c * BUCKETn;

    // async stage q (64x64 f16 contiguous = 512 x 16B) into LDS
    {
        const unsigned qb = lds_addr_of(qS);
#pragma unroll
        for (int c0 = tid; c0 < 512; c0 += 128)
            async_copy_b128(qb + (unsigned)c0 * 16u, sqk + baseQ * 64 + (size_t)c0 * 8);
    }
    // async stage k rows: own chunk then previous chunk (2 x 8KB contiguous)
    {
        const unsigned kb = lds_addr_of(kS);
#pragma unroll
        for (int c0 = tid; c0 < 1024; c0 += 128) {
            int half = c0 >> 9, off = c0 & 511;
            size_t srow = (size_t)b * NRn + (size_t)(half ? cp : c) * BUCKETn;
            async_copy_b128(kb + (unsigned)c0 * 16u, sqk + srow * 64 + (size_t)off * 8);
        }
    }
    // manual: v transposed scatter + time indices
    for (int idx = tid; idx < 128 * 64; idx += 128) {
        int j = idx >> 6, e = idx & 63;
        size_t srow = (size_t)b * NRn + (size_t)((j < 64) ? c : cp) * BUCKETn + (j & 63);
        vT[e * 128 + j] = sv[srow * 64 + e];
    }
    for (int i = tid; i < 64; i += 128) qtS[i] = sticker[baseQ + i] % Tsz;
    for (int j = tid; j < 128; j += 128) {
        size_t srow = (size_t)b * NRn + (size_t)((j < 64) ? c : cp) * BUCKETn + (j & 63);
        ktS[j] = sticker[srow] % Tsz;
    }
    wait_async0();
    __syncthreads();
    // normalize K rows (bk = qk / max(||qk||, 1e-12))
    for (int j = tid; j < 128; j += 128) {
        float ss = 0.f;
        for (int e = 0; e < 64; ++e) { float x = (float)kS[j * 64 + e]; ss += x * x; }
        float sc = 1.f / fmaxf(sqrtf(ss), 1e-12f);
        for (int e = 0; e < 64; ++e) kS[j * 64 + e] = (_Float16)((float)kS[j * 64 + e] * sc);
    }
    __syncthreads();
    // dots = q @ k^T : each wave does 16 rows x 128 cols
    const v8f vz = {0, 0, 0, 0, 0, 0, 0, 0};
    v8f acc[8];
#pragma unroll
    for (int j = 0; j < 8; ++j) acc[j] = vz;
#pragma unroll
    for (int kk = 0; kk < 64; kk += 32) {
        v16h a = fragA(qS + kk, 64, 16 * wave + (lane & 15), lane);
#pragma unroll
        for (int j = 0; j < 8; ++j) {
            v16h bf = fragB(kS + kk, 64, 16 * j, lane);
            acc[j] = __builtin_amdgcn_wmma_f32_16x16x32_f16(false, a, false, bf,
                                                            (short)0, acc[j], false, false);
        }
    }
    {
        const int nl = lane & 15, mhi = (lane & 16) ? 8 : 0;
#pragma unroll
        for (int j = 0; j < 8; ++j) {
            int ncol = 16 * j + nl;
            int ktj = ktS[ncol];
#pragma unroll
            for (int r = 0; r < 8; ++r) {
                int row = 16 * wave + mhi + r;
                int qti = qtS[row];
                float v = acc[j][r] * 0.125f;     // DH^-0.5
                if (qti < ktj) v = -1e9f;         // causal mask
                else if (qti == ktj) v = -5e4f;   // self mask
                dS[row * 128 + ncol] = v;
            }
        }
    }
    __syncthreads();
    // softmax per row; probs (f16) overwrite kS; lse -> global
    if (tid < 64) {
        int row = tid;
        float m = -3.4e38f;
        for (int j2 = 0; j2 < 128; ++j2) m = fmaxf(m, dS[row * 128 + j2]);
        float s = 0.f;
        for (int j2 = 0; j2 < 128; ++j2) s += expf(dS[row * 128 + j2] - m);
        float lse = m + logf(s);
        slse[baseQ + row] = lse;
        for (int j2 = 0; j2 < 128; ++j2)
            kS[row * 128 + j2] = (_Float16)expf(dS[row * 128 + j2] - lse);
    }
    __syncthreads();
    // so = probs @ v
    v8f acc2[4];
#pragma unroll
    for (int j = 0; j < 4; ++j) acc2[j] = vz;
#pragma unroll
    for (int kk = 0; kk < 128; kk += 32) {
        v16h a = fragA(kS + kk, 128, 16 * wave + (lane & 15), lane);
#pragma unroll
        for (int j = 0; j < 4; ++j) {
            v16h bf = fragB(vT + kk, 128, 16 * j, lane);
            acc2[j] = __builtin_amdgcn_wmma_f32_16x16x32_f16(false, a, false, bf,
                                                             (short)0, acc2[j], false, false);
        }
    }
    {
        const int nl = lane & 15, mhi = (lane & 16) ? 8 : 0;
#pragma unroll
        for (int j = 0; j < 4; ++j) {
#pragma unroll
            for (int r = 0; r < 8; ++r) {
                int row = 16 * wave + mhi + r;
                so[(baseQ + row) * 64 + 16 * j + nl] = acc2[j][r];
            }
        }
    }
}

// ---------------- unsort + logsumexp-weighted combine over hash rounds -------
__global__ void __launch_bounds__(256)
combine_unsort(const float* __restrict__ so, const float* __restrict__ slse,
               const int* __restrict__ undo, _Float16* __restrict__ attn16) {
    size_t gid = (size_t)blockIdx.x * 256 + threadIdx.x;    // over BH*T*DH
    int dh = (int)(gid & 63);
    size_t rem = gid >> 6;
    int t = (int)(rem % Tsz);
    int b = (int)(rem / Tsz);
    float l[NHASHn]; int pos[NHASHn];
    float m = -3.4e38f;
    for (int nh = 0; nh < NHASHn; ++nh) {
        pos[nh] = undo[(size_t)b * NRn + (size_t)nh * Tsz + t];
        l[nh] = slse[(size_t)b * NRn + pos[nh]];
        m = fmaxf(m, l[nh]);
    }
    float den = 0.f;
    for (int nh = 0; nh < NHASHn; ++nh) den += expf(l[nh] - m);
    float o = 0.f;
    for (int nh = 0; nh < NHASHn; ++nh)
        o += (expf(l[nh] - m) / den) * so[((size_t)b * NRn + pos[nh]) * 64 + dh];
    int bb = b / Hn, hh = b - bb * Hn;
    attn16[((size_t)bb * Tsz + t) * Esz + (size_t)hh * DHn + dh] = (_Float16)o;
}

// ---------------- mean pool of 0.5*(x1+x2) over T ----------------
__global__ void __launch_bounds__(256)
mean_pool(const float* __restrict__ x1, const float* __restrict__ x2,
          float* __restrict__ pooled) {
    int gid = blockIdx.x * 256 + threadIdx.x;   // over B*E
    if (gid >= Bsz * Esz) return;
    int bb = gid / Esz, e = gid - bb * Esz;
    float s = 0.f;
    for (int t = 0; t < Tsz; ++t) {
        size_t idx = ((size_t)bb * Tsz + t) * Esz + e;
        s += 0.5f * (x1[idx] + x2[idx]);
    }
    pooled[gid] = s * (1.f / Tsz);
}

// ---------------- final classifier ----------------
__global__ void final_proj(const float* __restrict__ pooled, const float* __restrict__ wf,
                           const float* __restrict__ bf, float* __restrict__ out) {
    int gid = threadIdx.x;
    if (gid >= Bsz * NCLASSn) return;
    int bb = gid / NCLASSn, cc = gid - bb * NCLASSn;
    float s = bf[cc];
    for (int e = 0; e < Esz; ++e) s += pooled[bb * Esz + e] * wf[e * NCLASSn + cc];
    out[gid] = s;
}

// =======================================================================
extern "C" void kernel_launch(void* const* d_in, const int* in_sizes, int n_in,
                              void* d_out, int out_size, void* d_ws, size_t ws_size,
                              hipStream_t stream) {
    const int*   x        = (const int*)d_in[0];
    const float* tok_emb  = (const float*)d_in[1];
    const float* pos_emb  = (const float*)d_in[2];
    const float* rotations= (const float*)d_in[3];
    const float* ln1_g    = (const float*)d_in[4];
    const float* ln1_b    = (const float*)d_in[5];
    const float* w_qk     = (const float*)d_in[6];
    const float* w_v      = (const float*)d_in[7];
    const float* w_o      = (const float*)d_in[8];
    const float* b_o      = (const float*)d_in[9];
    const float* ln2_g    = (const float*)d_in[10];
    const float* ln2_b    = (const float*)d_in[11];
    const float* w_ff1    = (const float*)d_in[12];
    const float* b_ff1    = (const float*)d_in[13];
    const float* w_ff2    = (const float*)d_in[14];
    const float* b_ff2    = (const float*)d_in[15];
    const float* w_final  = (const float*)d_in[16];
    const float* b_final  = (const float*)d_in[17];
    float* out = (float*)d_out;

    const size_t NTOK = (size_t)Bsz * Tsz;       // 8192
    const size_t NE   = NTOK * Esz;              // 4,194,304
    const size_t NROW = (size_t)BHn * NRn;       // 262,144

    // bump-allocate workspace (256B aligned)
    char* wsb = (char*)d_ws;
    size_t off = 0;
    auto alloc = [&](size_t bytes) -> char* {
        char* p = wsb + off;
        off += (bytes + 255) & ~(size_t)255;
        return p;
    };
    float*    x1     = (float*)alloc(NE * 4);
    float*    x2     = (float*)alloc(NE * 4);
    float*    qk32   = (float*)alloc(NE * 4);
    float*    v32    = (float*)alloc(NE * 4);
    float*    so32   = (float*)alloc(NROW * 64 * 4);
    float*    slse   = (float*)alloc(NROW * 4);
    float*    pooled = (float*)alloc((size_t)Bsz * Esz * 4);
    _Float16* ln16   = (_Float16*)alloc(NE * 2);
    _Float16* attn16 = (_Float16*)alloc(NE * 2);
    _Float16* ff16   = (_Float16*)alloc(NTOK * FFn * 2);
    _Float16* w16T   = (_Float16*)alloc((size_t)Esz * FFn * 2);   // B^T (NxK) f16
    _Float16* sqk16  = (_Float16*)alloc(NROW * 64 * 2);
    _Float16* sv16   = (_Float16*)alloc(NROW * 64 * 2);
    int*      keys   = (int*)alloc(NROW * 4);
    int*      sticker= (int*)alloc(NROW * 4);
    int*      undo   = (int*)alloc(NROW * 4);

    (void)in_sizes; (void)n_in; (void)out_size; (void)ws_size;
    (void)hipFuncSetAttribute((const void*)bitonic_sort,
                              hipFuncAttributeMaxDynamicSharedMemorySize, 131072);
    (void)hipFuncSetAttribute((const void*)lsh_attn_chunk,
                              hipFuncAttributeMaxDynamicSharedMemorySize, 74496);

    // embedding -> x1 = x2 = h
    embed_kernel<<<(int)NTOK, 256, 0, stream>>>(x, tok_emb, pos_emb, x1, x2);

    for (int d = 0; d < DEPTHn; ++d) {
        const float* l1g = ln1_g + (size_t)d * Esz;
        const float* l1b = ln1_b + (size_t)d * Esz;
        const float* l2g = ln2_g + (size_t)d * Esz;
        const float* l2b = ln2_b + (size_t)d * Esz;
        const float* wqk = w_qk + (size_t)d * Esz * Esz;
        const float* wv  = w_v  + (size_t)d * Esz * Esz;
        const float* wo  = w_o  + (size_t)d * Esz * Esz;
        const float* bo  = b_o  + (size_t)d * Esz;
        const float* wf1 = w_ff1 + (size_t)d * Esz * FFn;
        const float* bf1 = b_ff1 + (size_t)d * FFn;
        const float* wf2 = w_ff2 + (size_t)d * FFn * Esz;
        const float* bf2 = b_ff2 + (size_t)d * Esz;
        const float* rot = rotations + (size_t)d * DHn * NHASHn * (NBn / 2);

        // ---- attention branch: x1 += attn(ln(x2)) ----
        layernorm_to_f16<<<(int)NTOK, 256, 0, stream>>>(x2, l1g, l1b, ln16);

        cvt_transpose_f16<<<(Esz * Esz + 255) / 256, 256, 0, stream>>>(wqk, w16T, Esz, Esz);
        gemm_f16_wmma<<<dim3(Esz / 128, (int)(NTOK / 64)), 128, 0, stream>>>(
            ln16, w16T, qk32, nullptr, nullptr, nullptr, (int)NTOK, Esz, Esz, 0);

        cvt_transpose_f16<<<(Esz * Esz + 255) / 256, 256, 0, stream>>>(wv, w16T, Esz, Esz);
        gemm_f16_wmma<<<dim3(Esz / 128, (int)(NTOK / 64)), 128, 0, stream>>>(
            ln16, w16T, v32, nullptr, nullptr, nullptr, (int)NTOK, Esz, Esz, 0);

        lsh_hash<<<(int)(NROW / 256), 256, 0, stream>>>(qk32, rot, keys);
        bitonic_sort<<<BHn, 1024, 131072, stream>>>(keys, sticker, undo);
        gather_sorted<<<(int)(NROW / 4), 256, 0, stream>>>(sticker, qk32, v32, sqk16, sv16);
        lsh_attn_chunk<<<dim3(Cn, BHn), 128, 74496, stream>>>(sqk16, sv16, sticker, so32, slse);
        combine_unsort<<<(int)(NROW * 64 / 256), 256, 0, stream>>>(so32, slse, undo, attn16);

        cvt_transpose_f16<<<(Esz * Esz + 255) / 256, 256, 0, stream>>>(wo, w16T, Esz, Esz);
        gemm_f16_wmma<<<dim3(Esz / 128, (int)(NTOK / 64)), 128, 0, stream>>>(
            attn16, w16T, x1, nullptr, bo, x1, (int)NTOK, Esz, Esz, 0);

        // ---- FF branch: x2 += ff(ln(x1)) ----
        layernorm_to_f16<<<(int)NTOK, 256, 0, stream>>>(x1, l2g, l2b, ln16);

        cvt_transpose_f16<<<(Esz * FFn + 255) / 256, 256, 0, stream>>>(wf1, w16T, Esz, FFn);
        gemm_f16_wmma<<<dim3(FFn / 128, (int)(NTOK / 64)), 128, 0, stream>>>(
            ln16, w16T, nullptr, ff16, bf1, nullptr, (int)NTOK, FFn, Esz, 1);

        cvt_transpose_f16<<<(FFn * Esz + 255) / 256, 256, 0, stream>>>(wf2, w16T, FFn, Esz);
        gemm_f16_wmma<<<dim3(Esz / 128, (int)(NTOK / 64)), 128, 0, stream>>>(
            ff16, w16T, x2, nullptr, bf2, x2, (int)NTOK, Esz, FFn, 0);
    }

    mean_pool<<<(Bsz * Esz + 255) / 256, 256, 0, stream>>>(x1, x2, pooled);
    final_proj<<<1, 32, 0, stream>>>(pooled, w_final, b_final, out);
}
